// EncoderRNN_558345748990
// MI455X (gfx1250) — compile-verified, matrix operand order checked by reading, not run on hardware
//
#include <hip/hip_runtime.h>
#include <hip/hip_bf16.h>

// ---------------------------------------------------------------------------
// 3-layer GRU, B=4096, S=128, I=8, H=512.  Batch rows are independent through
// the recurrence: each workgroup owns BT=32 batch rows and keeps h resident in
// LDS (fp32 master + bf16 shadow) for all S*L = 384 recurrent GEMMs.
// Core op: [h | x] @ [Whh | Wih]^T via v_wmma_f32_16x16x32_bf16, K = 544
// (512 hidden + 8 input + 24 zero pad).  The input GEMM is fused into the
// WMMA stream for the r/z gates; the n gate keeps i_n separate (r scales only
// h_n), so it uses an 8-FMA scalar dot on the VALU pipe.
// Weights are pre-packed to bf16 in exact B-fragment tile order in d_ws.
// ---------------------------------------------------------------------------

typedef __attribute__((ext_vector_type(16))) __bf16 v16bf;
typedef __attribute__((ext_vector_type(8)))  float  v8f;

#define HID    512
#define SEQ    128
#define INP    8
#define NLAY   3
#define BT     32           // batch rows per workgroup
#define HB_STR 552          // bf16 h row stride: 512 h + 8 x + 24 zeros + 8 pad
#define HF_STR 516          // f32  h row stride
#define G3H    (3*HID)      // 1536
#define KBLK   17           // K blocks of 32: 16 for h, 1 for fused x

union FragB {
    v16bf v;
    uint4 q[2];
};

// Pack weights -> bf16 tiles in exact WMMA B-fragment order.
// dst layout: [l][rb 0..95][kb 0..16][lane 0..31][elem 0..15]
//   kb < 16 : Whh[l][rb*16 + lane%16][kb*32 + kl]
//   kb ==16 : Wih[l][row][kl] for kl<8 and r/z rows (row<1024), else 0
// kl = (e&7) + (lane>=16 ? 8 : 0) + (e&8 ? 16 : 0)
__global__ void gru_pack_w(const float* __restrict__ whh,
                           const float* __restrict__ wih,
                           __bf16* __restrict__ dst, int n)
{
    int idx = blockIdx.x * blockDim.x + threadIdx.x;
    if (idx >= n) return;
    int e    = idx & 15;
    int lane = (idx >> 4) & 31;
    int kb   = (idx >> 9) % KBLK;
    int rb   = (idx / (KBLK << 9)) % 96;
    int l    = idx / (96 * KBLK * 512);
    int lc = lane & 15, hi = lane >> 4;
    int kl = (e & 7) + hi * 8 + ((e & 8) ? 16 : 0);
    int r  = rb * 16 + lc;
    float v;
    if (kb < 16) {
        v = whh[((size_t)l * G3H + r) * HID + kb * 32 + kl];
    } else {
        v = (kl < INP && r < 2 * HID) ? wih[((size_t)l * G3H + r) * INP + kl] : 0.0f;
    }
    dst[idx] = (__bf16)v;
}

__global__ __launch_bounds__(256, 1)
void gru_recurrent(const float* __restrict__ input,   // [B][S][I]
                   const float* __restrict__ wih,     // [L][1536][8]
                   const float* __restrict__ bih,     // [L][1536]
                   const float* __restrict__ bhh,     // [L][1536]
                   const __bf16* __restrict__ w_t,    // packed tiles
                   float* __restrict__ out)           // [B][512]
{
    extern __shared__ char smem[];
    float*  xs    = (float*)smem;                     //  1024 B : x_t tile [32][8] f32
    float*  bih_s = xs + BT * INP;                    // 18432 B
    float*  bhh_s = bih_s + NLAY * G3H;               // 18432 B
    float*  hf    = bhh_s + NLAY * G3H;               // 66048 B : fp32 master h
    __bf16* wn_s  = (__bf16*)(hf + BT * HF_STR);      // 24576 B : Wih n-gate slice bf16
    __bf16* hb    = wn_s + NLAY * HID * INP;          // 70656 B : bf16 [h|x|0], double buf

    const int tid   = threadIdx.x;
    const int wave  = tid >> 5;
    const int lane  = tid & 31;
    const int lcol  = lane & 15;
    const int lhalf = (lane < 16) ? 0 : 8;
    const int Bbase = blockIdx.x * BT;

    // Stage n-gate Wih + biases, zero h (covers the K 520..543 zero region too)
    for (int i = tid; i < NLAY * HID * INP; i += 256) {
        int l = i / (HID * INP), rest = i % (HID * INP);
        wn_s[i] = (__bf16)wih[((size_t)l * G3H + 2 * HID) * INP + rest];
    }
    for (int i = tid; i < NLAY * G3H; i += 256) { bih_s[i] = bih[i]; bhh_s[i] = bhh[i]; }
    for (int i = tid; i < BT * HF_STR; i += 256) hf[i] = 0.0f;
    for (int i = tid; i < 2 * BT * HB_STR; i += 256) hb[i] = (__bf16)0.0f;
    __syncthreads();

    int cur = 0;
    for (int t = 0; t < SEQ; ++t) {
        // x_t -> LDS f32 (for n-gate dot) and bf16 into BOTH h buffers' K 512..519
        {
            int r = tid >> 3, c = tid & 7;
            float xv = input[(size_t)(Bbase + r) * SEQ * INP + t * INP + c];
            xs[r * INP + c] = xv;
            __bf16 xb = (__bf16)xv;
            hb[r * HB_STR + HID + c] = xb;
            hb[BT * HB_STR + r * HB_STR + HID + c] = xb;
        }
        __syncthreads();

        for (int l = 0; l < NLAY; ++l) {
            const __bf16* hbc = hb + cur * BT * HB_STR;          // GEMM input
            __bf16*       hbn = hb + (cur ^ 1) * BT * HB_STR;    // next-state shadow

            const size_t rstep = (size_t)KBLK * 512;             // elems per rb block
            const size_t lbase = (size_t)l * 96 * rstep;

            // this wave owns hidden columns [wave*64, wave*64+64)
            for (int g = 0; g < 4; ++g) {
                const int cb   = wave * 64 + g * 16;
                const int colH = cb + lcol;
                const size_t rb_r = lbase + (size_t)((0 * HID + cb) >> 4) * rstep;
                const size_t rb_z = lbase + (size_t)((1 * HID + cb) >> 4) * rstep;
                const size_t rb_n = lbase + (size_t)((2 * HID + cb) >> 4) * rstep;

                v8f acc0 = {}, acc1 = {}, acc2 = {}, acc3 = {}, acc4 = {}, acc5 = {};

                // K = 512 over h for all 3 gates
                for (int kb = 0; kb < 16; ++kb) {
                    FragB a0, a1, b0, b1, b2;
                    const __bf16* ar0 = hbc + (size_t)lcol * HB_STR + kb * 32 + lhalf;
                    const __bf16* ar1 = hbc + (size_t)(16 + lcol) * HB_STR + kb * 32 + lhalf;
                    a0.q[0] = *(const uint4*)(ar0);
                    a0.q[1] = *(const uint4*)(ar0 + 16);
                    a1.q[0] = *(const uint4*)(ar1);
                    a1.q[1] = *(const uint4*)(ar1 + 16);

                    const size_t ko = (size_t)kb * 512 + (size_t)lane * 16;
                    const __bf16* tp0 = w_t + rb_r + ko;
                    const __bf16* tp1 = w_t + rb_z + ko;
                    const __bf16* tp2 = w_t + rb_n + ko;
                    b0.q[0] = *(const uint4*)(tp0); b0.q[1] = *(const uint4*)(tp0 + 8);
                    b1.q[0] = *(const uint4*)(tp1); b1.q[1] = *(const uint4*)(tp1 + 8);
                    b2.q[0] = *(const uint4*)(tp2); b2.q[1] = *(const uint4*)(tp2 + 8);

                    acc0 = __builtin_amdgcn_wmma_f32_16x16x32_bf16(false, a0.v, false, b0.v, (short)0, acc0, false, false);
                    acc1 = __builtin_amdgcn_wmma_f32_16x16x32_bf16(false, a1.v, false, b0.v, (short)0, acc1, false, false);
                    acc2 = __builtin_amdgcn_wmma_f32_16x16x32_bf16(false, a0.v, false, b1.v, (short)0, acc2, false, false);
                    acc3 = __builtin_amdgcn_wmma_f32_16x16x32_bf16(false, a1.v, false, b1.v, (short)0, acc3, false, false);
                    acc4 = __builtin_amdgcn_wmma_f32_16x16x32_bf16(false, a0.v, false, b2.v, (short)0, acc4, false, false);
                    acc5 = __builtin_amdgcn_wmma_f32_16x16x32_bf16(false, a1.v, false, b2.v, (short)0, acc5, false, false);
                }
                // Fused K block 16: x @ Wih for r and z gates only
                {
                    FragB a0, a1, b0, b1;
                    const __bf16* ar0 = hbc + (size_t)lcol * HB_STR + HID + lhalf;
                    const __bf16* ar1 = hbc + (size_t)(16 + lcol) * HB_STR + HID + lhalf;
                    a0.q[0] = *(const uint4*)(ar0);
                    a0.q[1] = *(const uint4*)(ar0 + 16);
                    a1.q[0] = *(const uint4*)(ar1);
                    a1.q[1] = *(const uint4*)(ar1 + 16);

                    const size_t ko = (size_t)16 * 512 + (size_t)lane * 16;
                    const __bf16* tp0 = w_t + rb_r + ko;
                    const __bf16* tp1 = w_t + rb_z + ko;
                    b0.q[0] = *(const uint4*)(tp0); b0.q[1] = *(const uint4*)(tp0 + 8);
                    b1.q[0] = *(const uint4*)(tp1); b1.q[1] = *(const uint4*)(tp1 + 8);

                    acc0 = __builtin_amdgcn_wmma_f32_16x16x32_bf16(false, a0.v, false, b0.v, (short)0, acc0, false, false);
                    acc1 = __builtin_amdgcn_wmma_f32_16x16x32_bf16(false, a1.v, false, b0.v, (short)0, acc1, false, false);
                    acc2 = __builtin_amdgcn_wmma_f32_16x16x32_bf16(false, a0.v, false, b1.v, (short)0, acc2, false, false);
                    acc3 = __builtin_amdgcn_wmma_f32_16x16x32_bf16(false, a1.v, false, b1.v, (short)0, acc3, false, false);
                }

                // Gate math. C/D layout: lane L, VGPR i -> row i + 8*(L>=16), col L%16.
                float wn[INP];
                #pragma unroll
                for (int k = 0; k < INP; ++k)
                    wn[k] = (float)wn_s[((size_t)l * HID + colH) * INP + k];
                const float brz = bih_s[l * G3H + colH]       + bhh_s[l * G3H + colH];
                const float bzz = bih_s[l * G3H + HID + colH] + bhh_s[l * G3H + HID + colH];
                const float bni = bih_s[l * G3H + 2 * HID + colH];
                const float bnh = bhh_s[l * G3H + 2 * HID + colH];

                #pragma unroll
                for (int mt = 0; mt < 2; ++mt) {
                    #pragma unroll
                    for (int i = 0; i < 8; ++i) {
                        const int m = mt * 16 + lhalf + i;
                        float gn = bni;
                        const float* xr = xs + m * INP;
                        #pragma unroll
                        for (int k = 0; k < INP; ++k)
                            gn = fmaf(xr[k], wn[k], gn);
                        const float rs = (mt ? acc1[i] : acc0[i]) + brz;   // i_r + h_r
                        const float zs = (mt ? acc3[i] : acc2[i]) + bzz;   // i_z + h_z
                        const float hn = (mt ? acc5[i] : acc4[i]) + bnh;   // h_n only
                        const float rg = 1.0f / (1.0f + __expf(-rs));
                        const float zg = 1.0f / (1.0f + __expf(-zs));
                        const float ng = tanhf(gn + rg * hn);
                        const float ho = hf[m * HF_STR + colH];
                        const float hv = (1.0f - zg) * ng + zg * ho;
                        hf[m * HF_STR + colH]  = hv;          // fp32 master (unique owner)
                        hbn[m * HB_STR + colH] = (__bf16)hv;  // bf16 shadow for next GEMM
                    }
                }
            }
            __syncthreads();   // all waves done with layer l before buffers flip
            cur ^= 1;
        }
    }

    __syncthreads();
    for (int i = tid; i < BT * HID; i += 256) {
        int r = i >> 9, c = i & 511;
        out[(size_t)(Bbase + r) * HID + c] = hf[r * HF_STR + c];
    }
}

extern "C" void kernel_launch(void* const* d_in, const int* in_sizes, int n_in,
                              void* d_out, int out_size, void* d_ws, size_t ws_size,
                              hipStream_t stream) {
    const float* input = (const float*)d_in[0];   // [4096][128][8]
    const float* wih   = (const float*)d_in[1];   // [3][1536][8]
    const float* whh   = (const float*)d_in[2];   // [3][1536][512]
    const float* bihp  = (const float*)d_in[3];   // [3][1536]
    const float* bhhp  = (const float*)d_in[4];   // [3][1536]
    float* out = (float*)d_out;
    __bf16* w_t = (__bf16*)d_ws;                  // ~4.8 MB packed bf16 tiles

    const int n = NLAY * 96 * KBLK * 512;
    gru_pack_w<<<(n + 255) / 256, 256, 0, stream>>>(whh, wih, w_t, n);

    const size_t smem =
        (size_t)BT * INP * 4 +            // xs
        (size_t)NLAY * G3H * 4 * 2 +      // bih_s + bhh_s
        (size_t)BT * HF_STR * 4 +         // hf
        (size_t)NLAY * HID * INP * 2 +    // wn_s (bf16, n-gate only)
        (size_t)2 * BT * HB_STR * 2;      // hb double buffer (bf16)

    gru_recurrent<<<4096 / BT, 256, smem, stream>>>(input, wih, bihp, bhhp, w_t, out);
}
